// SpatialAttention_6141803233366
// MI455X (gfx1250) — compile-verified, matrix operand order checked by reading, not run on hardware
//
#include <hip/hip_runtime.h>

// ---------------------------------------------------------------------------
// Types & helpers
// ---------------------------------------------------------------------------
typedef __attribute__((ext_vector_type(16))) __bf16 v16bf;
typedef __attribute__((ext_vector_type(8)))  float  v8f;

struct __align__(16) U4 { unsigned int x, y, z, w; };

union Frag {
  v16bf v;
  unsigned int u[8];
  U4 q[2];
};

__device__ __forceinline__ unsigned short f32_to_bf16(float f) {
  union { float f; unsigned int u; } cv; cv.f = f;
  unsigned int u = cv.u;
  u += 0x7FFFu + ((u >> 16) & 1u);           // round-to-nearest-even
  return (unsigned short)(u >> 16);
}
__device__ __forceinline__ v8f wmma_bf16(const Frag& a, const Frag& b, v8f c) {
  return __builtin_amdgcn_wmma_f32_16x16x32_bf16(false, a.v, false, b.v,
                                                 (short)0, c, false, false);
}
// 50*tanh(s*0.125/50) via fast exp:  50 - 100/(exp(0.005*s)+1)
__device__ __forceinline__ float softcap(float s) {
  const float e = __expf(s * 0.005f);
  return 50.0f - 100.0f / (e + 1.0f);
}

#define DIMC   1024
#define NH_C   16
#define NKV_C  4
#define HD_C   64
#define S_C    264
#define BT_C   64
#define NROWS  (BT_C * S_C)            // 16896 (multiple of 16)
#define NMB    (NROWS / 16)            // 1056 row blocks
#define LN1E4_OVER_Q 0.5756462732485115f   // ln(10000)/16

// ---------------------------------------------------------------------------
// K0: fp32 -> bf16 conversion (weights and activations; bandwidth-bound)
// ---------------------------------------------------------------------------
__global__ void cvt_bf16_kernel(const float* __restrict__ src,
                                unsigned short* __restrict__ dst, int n) {
  int i = blockIdx.x * 256 + threadIdx.x;
  if (i < n) dst[i] = f32_to_bf16(src[i]);
}

// ---------------------------------------------------------------------------
// K1: fused QKV projection + QKNorm + partial 2D RoPE
// One wave = 16 tokens x 64 cols (one head) of one of 24 units
// (16 Q heads, 4 K heads, 4 V heads), K = 1024 via 32 WMMA k-steps.
// Pure b128-load -> WMMA inner loop (all operands pre-converted bf16).
// ---------------------------------------------------------------------------
__global__ __launch_bounds__(256) void qkv_kernel(
    const unsigned short* __restrict__ x_bf,
    const unsigned short* __restrict__ wq_bf,
    const unsigned short* __restrict__ wk_bf,
    const unsigned short* __restrict__ wv_bf,
    const float* __restrict__ qnw,
    const float* __restrict__ knw,
    unsigned short* __restrict__ q_bf,   // (BT,NH,S,HD)
    unsigned short* __restrict__ k_bf,   // (BT,NKV,S,HD)
    unsigned short* __restrict__ v_t)    // (BT,NKV,HD,S) transposed
{
  const int lane  = threadIdx.x & 31;
  const int wave  = threadIdx.x >> 5;
  const int l16   = lane & 15;
  const int half  = lane >> 4;
  const int task  = blockIdx.x * 8 + wave;
  const int unit  = task % 24;
  const int mb    = task / 24;

  const unsigned short* wptr;
  if (unit < 16)      wptr = wq_bf + (size_t)unit * (HD_C * DIMC);
  else if (unit < 20) wptr = wk_bf + (size_t)(unit - 16) * (HD_C * DIMC);
  else                wptr = wv_bf + (size_t)(unit - 20) * (HD_C * DIMC);

  const int rowA = mb * 16 + l16;                 // A-operand row = lane16
  const unsigned short* aptr = x_bf + (size_t)rowA * DIMC;

  v8f acc[4] = {};

  for (int k0 = 0; k0 < DIMC; k0 += 32) {
    Frag a;
    a.q[0] = *(const U4*)(aptr + k0 + 8 * half);
    a.q[1] = *(const U4*)(aptr + k0 + 16 + 8 * half);
#pragma unroll
    for (int t = 0; t < 4; ++t) {
      Frag b;
      const unsigned short* wp = wptr + (size_t)(t * 16 + l16) * DIMC + k0;
      b.q[0] = *(const U4*)(wp + 8 * half);
      b.q[1] = *(const U4*)(wp + 16 + 8 * half);
      acc[t] = wmma_bf16(a, b, acc[t]);
    }
  }

  const bool isV = (unit >= 20);
  const float* nw = (unit < 16) ? qnw : knw;
  float wcol[4];
  if (!isV) {
#pragma unroll
    for (int t = 0; t < 4; ++t) wcol[t] = nw[t * 16 + l16];
  }
  const float invf = __expf(-(float)l16 * LN1E4_OVER_Q);

#pragma unroll
  for (int j = 0; j < 8; ++j) {
    const int gr = mb * 16 + j + 8 * half;        // C-layout row for this lane
    const int bt = gr / S_C;
    const int s  = gr - bt * S_C;
    float c0 = acc[0][j], c1 = acc[1][j], c2 = acc[2][j], c3 = acc[3][j];

    if (!isV) {
      // RMSNorm across head_dim (cross-lane over the 16-lane half)
      float ss = c0 * c0 + c1 * c1 + c2 * c2 + c3 * c3;
#pragma unroll
      for (int m = 1; m < 16; m <<= 1) ss += __shfl_xor(ss, m, 32);
      const float r = rsqrtf(ss * (1.0f / 64.0f) + 1e-6f);
      c0 *= r * wcol[0]; c1 *= r * wcol[1]; c2 *= r * wcol[2]; c3 *= r * wcol[3];
      if (s < 256) {                               // partial 2D RoPE
        const float prow = (float)(s >> 4);
        const float pcol = (float)(s & 15);
        const float a0 = prow * invf, a1 = pcol * invf;
        const float cs0 = __cosf(a0), sn0 = __sinf(a0);
        const float cs1 = __cosf(a1), sn1 = __sinf(a1);
        const float n0 = c0 * cs0 - c2 * sn0;
        const float n1 = c1 * cs1 - c3 * sn1;
        const float n2 = c2 * cs0 + c0 * sn0;
        const float n3 = c3 * cs1 + c1 * sn1;
        c0 = n0; c1 = n1; c2 = n2; c3 = n3;
      }
    }

    if (unit < 16) {
      unsigned short* dst =
          q_bf + ((size_t)(bt * NH_C + unit) * S_C + s) * HD_C + l16;
      dst[0]  = f32_to_bf16(c0); dst[16] = f32_to_bf16(c1);
      dst[32] = f32_to_bf16(c2); dst[48] = f32_to_bf16(c3);
    } else if (unit < 20) {
      unsigned short* dst =
          k_bf + ((size_t)(bt * NKV_C + (unit - 16)) * S_C + s) * HD_C + l16;
      dst[0]  = f32_to_bf16(c0); dst[16] = f32_to_bf16(c1);
      dst[32] = f32_to_bf16(c2); dst[48] = f32_to_bf16(c3);
    } else {
      unsigned short* base =
          v_t + (size_t)(bt * NKV_C + (unit - 20)) * (HD_C * S_C);
      base[(size_t)(l16 +  0) * S_C + s] = f32_to_bf16(c0);
      base[(size_t)(l16 + 16) * S_C + s] = f32_to_bf16(c1);
      base[(size_t)(l16 + 32) * S_C + s] = f32_to_bf16(c2);
      base[(size_t)(l16 + 48) * S_C + s] = f32_to_bf16(c3);
    }
  }
}

// ---------------------------------------------------------------------------
// K2: GQA flash attention with tanh soft-cap.
// One wave = (bt, head, 16-query block). 9 key chunks of 32.
// ---------------------------------------------------------------------------
__global__ __launch_bounds__(128) void attn_kernel(
    const unsigned short* __restrict__ q_bf,
    const unsigned short* __restrict__ k_bf,
    const unsigned short* __restrict__ v_t,
    unsigned short* __restrict__ attn)          // (BT,S,NH*HD) bf16
{
  __shared__ __align__(16) unsigned short pshare[4][16 * 32];

  const int lane = threadIdx.x & 31;
  const int wave = threadIdx.x >> 5;
  const int l16  = lane & 15;
  const int half = lane >> 4;

  const int task = blockIdx.x * 4 + wave;
  const int qb   = task % 17;
  const int head = (task / 17) % NH_C;
  const int bt   = task / (17 * NH_C);
  const int kv   = head >> 2;

  const unsigned short* qbase = q_bf + (size_t)(bt * NH_C + head) * S_C * HD_C;
  const unsigned short* kbase = k_bf + (size_t)(bt * NKV_C + kv) * S_C * HD_C;
  const unsigned short* vbase = v_t  + (size_t)(bt * NKV_C + kv) * HD_C * S_C;

  // Resident Q fragments (A operand, 16x64 = two k-steps)
  Frag qa[2];
  const int qrow = qb * 16 + l16;
  const bool qok = qrow < S_C;
#pragma unroll
  for (int ks = 0; ks < 2; ++ks) {
    if (qok) {
      const unsigned short* p = qbase + (size_t)qrow * HD_C + ks * 32;
      qa[ks].q[0] = *(const U4*)(p + 8 * half);
      qa[ks].q[1] = *(const U4*)(p + 16 + 8 * half);
    } else {
#pragma unroll
      for (int i = 0; i < 8; ++i) qa[ks].u[i] = 0;
    }
  }

  v8f acc[4] = {};
  float rowm[8], rowl[8];
#pragma unroll
  for (int j = 0; j < 8; ++j) { rowm[j] = -1e30f; rowl[j] = 0.0f; }

  unsigned short* pl = pshare[wave];

  for (int ch = 0; ch < 9; ++ch) {
    const int kchunk = ch * 32;
    v8f st[2];
#pragma unroll
    for (int t = 0; t < 2; ++t) {
      const int key = kchunk + t * 16 + l16;
      const bool kok = key < S_C;
      Frag bk0, bk1;
      if (kok) {
        const unsigned short* kp = kbase + (size_t)key * HD_C;
        bk0.q[0] = *(const U4*)(kp +       8 * half);
        bk0.q[1] = *(const U4*)(kp + 16 +  8 * half);
        bk1.q[0] = *(const U4*)(kp + 32 +  8 * half);
        bk1.q[1] = *(const U4*)(kp + 48 +  8 * half);
      } else {
#pragma unroll
        for (int i = 0; i < 8; ++i) { bk0.u[i] = 0; bk1.u[i] = 0; }
      }
      v8f s = {};
      s = wmma_bf16(qa[0], bk0, s);
      s = wmma_bf16(qa[1], bk1, s);
#pragma unroll
      for (int j = 0; j < 8; ++j) {
        s[j] = kok ? softcap(s[j]) : -1e30f;       // scale+cap fused in softcap
      }
      st[t] = s;
    }

    // Online softmax update (per C-layout row j; reductions across 16 lanes)
    float pf0[8], pf1[8];
#pragma unroll
    for (int j = 0; j < 8; ++j) {
      float mx = fmaxf(st[0][j], st[1][j]);
#pragma unroll
      for (int m = 1; m < 16; m <<= 1) mx = fmaxf(mx, __shfl_xor(mx, m, 32));
      const float newm = fmaxf(rowm[j], mx);
      const float sf = __expf(rowm[j] - newm);
      const float p0 = __expf(st[0][j] - newm);
      const float p1 = __expf(st[1][j] - newm);
      float ps = p0 + p1;
#pragma unroll
      for (int m = 1; m < 16; m <<= 1) ps += __shfl_xor(ps, m, 32);
      rowl[j] = rowl[j] * sf + ps;
      rowm[j] = newm;
      pf0[j] = p0; pf1[j] = p1;
#pragma unroll
      for (int t = 0; t < 4; ++t) acc[t][j] *= sf;
    }

    // P (16x32) via LDS: C-layout store, A-layout reload
#pragma unroll
    for (int j = 0; j < 8; ++j) {
      pl[(j + 8 * half) * 32 +      l16] = f32_to_bf16(pf0[j]);
      pl[(j + 8 * half) * 32 + 16 + l16] = f32_to_bf16(pf1[j]);
    }
    asm volatile("s_wait_dscnt 0" ::: "memory");
    Frag pa;
    {
      const unsigned short* pp = pl + l16 * 32 + 8 * half;
      pa.q[0] = *(const U4*)(pp);
      pa.q[1] = *(const U4*)(pp + 16);
    }

    // acc += P @ V   (B operand from transposed V: contiguous key runs)
#pragma unroll
    for (int n = 0; n < 4; ++n) {
      Frag bv;
      const unsigned short* vp = vbase + (size_t)(n * 16 + l16) * S_C + kchunk;
      const int k0 = kchunk + 8 * half;
      const int k1 = kchunk + 16 + 8 * half;
      if (k0 < S_C) bv.q[0] = *(const U4*)(vp + 8 * half);
      else { bv.u[0] = 0; bv.u[1] = 0; bv.u[2] = 0; bv.u[3] = 0; }
      if (k1 < S_C) bv.q[1] = *(const U4*)(vp + 16 + 8 * half);
      else { bv.u[4] = 0; bv.u[5] = 0; bv.u[6] = 0; bv.u[7] = 0; }
      acc[n] = wmma_bf16(pa, bv, acc[n]);
    }
  }

#pragma unroll
  for (int j = 0; j < 8; ++j) {
    const int q = qb * 16 + j + 8 * half;
    if (q < S_C) {
      const float inv = 1.0f / rowl[j];
      unsigned short* dst =
          attn + ((size_t)(bt * S_C + q)) * (NH_C * HD_C) + head * HD_C + l16;
      dst[0]  = f32_to_bf16(acc[0][j] * inv);
      dst[16] = f32_to_bf16(acc[1][j] * inv);
      dst[32] = f32_to_bf16(acc[2][j] * inv);
      dst[48] = f32_to_bf16(acc[3][j] * inv);
    }
  }
}

// ---------------------------------------------------------------------------
// K3: output projection  out = attn @ wo^T  (fp32 result)
// ---------------------------------------------------------------------------
__global__ __launch_bounds__(256) void oproj_kernel(
    const unsigned short* __restrict__ attn,
    const unsigned short* __restrict__ wo_bf,
    float* __restrict__ out)
{
  const int lane = threadIdx.x & 31;
  const int wave = threadIdx.x >> 5;
  const int l16  = lane & 15;
  const int half = lane >> 4;
  const int task = blockIdx.x * 8 + wave;
  const int nb   = task % 16;     // 16 column blocks of 64
  const int mb   = task / 16;     // 1056 row blocks

  const unsigned short* ap = attn + (size_t)(mb * 16 + l16) * DIMC;
  const unsigned short* wbase = wo_bf + (size_t)nb * (HD_C * DIMC);

  v8f acc[4] = {};
  for (int k0 = 0; k0 < DIMC; k0 += 32) {
    Frag a;
    a.q[0] = *(const U4*)(ap + k0 + 8 * half);
    a.q[1] = *(const U4*)(ap + k0 + 16 + 8 * half);
#pragma unroll
    for (int t = 0; t < 4; ++t) {
      Frag b;
      const unsigned short* wp = wbase + (size_t)(t * 16 + l16) * DIMC + k0;
      b.q[0] = *(const U4*)(wp + 8 * half);
      b.q[1] = *(const U4*)(wp + 16 + 8 * half);
      acc[t] = wmma_bf16(a, b, acc[t]);
    }
  }
#pragma unroll
  for (int j = 0; j < 8; ++j) {
    const int row = mb * 16 + j + 8 * half;
    float* orow = out + (size_t)row * DIMC + nb * 64 + l16;
    orow[0]  = acc[0][j];
    orow[16] = acc[1][j];
    orow[32] = acc[2][j];
    orow[48] = acc[3][j];
  }
}

// ---------------------------------------------------------------------------
// Host launcher
// ---------------------------------------------------------------------------
extern "C" void kernel_launch(void* const* d_in, const int* in_sizes, int n_in,
                              void* d_out, int out_size, void* d_ws, size_t ws_size,
                              hipStream_t stream) {
  (void)in_sizes; (void)n_in; (void)out_size; (void)ws_size;
  const float* x   = (const float*)d_in[0];
  const float* wq  = (const float*)d_in[1];
  const float* wk  = (const float*)d_in[2];
  const float* wv  = (const float*)d_in[3];
  const float* wo  = (const float*)d_in[4];
  const float* qnw = (const float*)d_in[5];
  const float* knw = (const float*)d_in[6];
  float* out = (float*)d_out;

  char* ws = (char*)d_ws;
  size_t off = 0;
  auto wsAlloc = [&](size_t bytes) -> void* {
    void* p = ws + off;
    off += (bytes + 255) & ~(size_t)255;
    return p;
  };
  const size_t nWQ = (size_t)NH_C  * HD_C * DIMC;    // 1M
  const size_t nWK = (size_t)NKV_C * HD_C * DIMC;    // 256K
  const size_t nWO = (size_t)DIMC  * NH_C * HD_C;    // 1M
  const size_t nX  = (size_t)NROWS * DIMC;           // 17.3M
  unsigned short* wq_bf  = (unsigned short*)wsAlloc(nWQ * 2);
  unsigned short* wk_bf  = (unsigned short*)wsAlloc(nWK * 2);
  unsigned short* wv_bf  = (unsigned short*)wsAlloc(nWK * 2);
  unsigned short* wo_bf  = (unsigned short*)wsAlloc(nWO * 2);
  unsigned short* x_bf   = (unsigned short*)wsAlloc(nX * 2);
  unsigned short* q_bf   = (unsigned short*)wsAlloc((size_t)BT_C * NH_C  * S_C * HD_C * 2);
  unsigned short* k_bf   = (unsigned short*)wsAlloc((size_t)BT_C * NKV_C * S_C * HD_C * 2);
  unsigned short* v_t    = (unsigned short*)wsAlloc((size_t)BT_C * NKV_C * HD_C * S_C * 2 + 256);
  unsigned short* attn_b = (unsigned short*)wsAlloc(nX * 2);

  cvt_bf16_kernel<<<(int)((nWQ + 255) / 256), 256, 0, stream>>>(wq, wq_bf, (int)nWQ);
  cvt_bf16_kernel<<<(int)((nWK + 255) / 256), 256, 0, stream>>>(wk, wk_bf, (int)nWK);
  cvt_bf16_kernel<<<(int)((nWK + 255) / 256), 256, 0, stream>>>(wv, wv_bf, (int)nWK);
  cvt_bf16_kernel<<<(int)((nWO + 255) / 256), 256, 0, stream>>>(wo, wo_bf, (int)nWO);
  cvt_bf16_kernel<<<(int)((nX  + 255) / 256), 256, 0, stream>>>(x,  x_bf,  (int)nX);

  // K1: 1056 row-blocks x 24 units, 8 waves per block
  qkv_kernel<<<(NMB * 24) / 8, 256, 0, stream>>>(x_bf, wq_bf, wk_bf, wv_bf,
                                                 qnw, knw, q_bf, k_bf, v_t);
  // K2: 64 bt x 16 heads x 17 q-blocks, 4 waves per block
  attn_kernel<<<(BT_C * NH_C * 17) / 4, 128, 0, stream>>>(q_bf, k_bf, v_t, attn_b);
  // K3: 1056 row-blocks x 16 col-blocks, 8 waves per block
  oproj_kernel<<<(NMB * 16) / 8, 256, 0, stream>>>(attn_b, wo_bf, out);
}
